// FlashLlamaAttention_90048284328533
// MI455X (gfx1250) — compile-verified, matrix operand order checked by reading, not run on hardware
//
#include <hip/hip_runtime.h>

// ---------------------------------------------------------------------------
// Problem constants (from the reference): B=4, S=1024, E=4096, H=32, KV=8,
// D=128, G=4. All GEMMs run on v_wmma_f32_16x16x32_bf16 (wave32, CDNA5),
// with Tensor-Data-Mover staging of block tiles into LDS.
// ---------------------------------------------------------------------------
#define B_ 4
#define S_ 1024
#define E_ 4096
#define H_ 32
#define KV_ 8
#define D_ 128
#define M_ (B_ * S_)          // 4096 token rows

typedef __attribute__((ext_vector_type(16))) __bf16 v16bf;
typedef __attribute__((ext_vector_type(8)))  float  v8f;
typedef __attribute__((ext_vector_type(4)))  unsigned int v4u;
typedef __attribute__((ext_vector_type(8)))  unsigned int v8u;

union Frag {                    // one WMMA 16x32 bf16 operand per lane: 32 bytes
    v16bf v;
    uint4 u4[2];
};

__device__ __forceinline__ unsigned short f2bf(float f) {
    unsigned int u = __builtin_bit_cast(unsigned int, f);
    unsigned int r = (u + 0x7FFFu + ((u >> 16) & 1u)) >> 16;
    return (unsigned short)r;
}
__device__ __forceinline__ float bf2f(unsigned short h) {
    unsigned int u = ((unsigned int)h) << 16;
    return __builtin_bit_cast(float, u);
}

// ---------------------------------------------------------------------------
// TDM: issue a 2D bf16 tile load Global->LDS via tensor_load_to_lds.
// D# group0/group1 packed per CDNA5 ISA 8.3/8.4:
//   g0: count=1 | lds_addr | global_addr[56:0] | type=2
//   g1: data_size=2B, pad_enable, pad_interval=32 DWORDs, pad_amount=4 DWORDs
//       tensor_dim0=K, tensor_dim1=rows, tile_dim0=tile_k, tile_dim1=rows,
//       tensor_dim0_stride=K (elements)
// Padded LDS row pitch: tile_k*2 + 16 bytes (keeps ds_load_b128 16B-aligned
// and staggers banks).
// ---------------------------------------------------------------------------
__device__ __forceinline__ void tdm_load_2d_bf16(unsigned lds_off,
                                                 const unsigned short* gptr,
                                                 unsigned K, unsigned rows,
                                                 unsigned tile_k) {
    unsigned long long ga = (unsigned long long)(size_t)gptr;
    v4u g0;
    g0[0] = 1u;                                        // count=1, user mode
    g0[1] = lds_off;                                   // LDS byte address
    g0[2] = (unsigned)ga;                              // global_addr[31:0]
    g0[3] = (unsigned)((ga >> 32) & 0x1FFFFFFu) | (2u << 30);  // [56:32] | type=2
    v8u g1;
    g1[0] = 0x07110000u;        // data_size=1(2B) | pad_enable | intv=32dw | amt=4dw
    g1[1] = (K & 0xFFFFu) << 16;                       // tensor_dim0 lo16
    g1[2] = (K >> 16) | ((rows & 0xFFFFu) << 16);      // dim0 hi16 | dim1 lo16
    g1[3] = (rows >> 16) | (tile_k << 16);             // dim1 hi16 | tile_dim0
    g1[4] = rows & 0xFFFFu;                            // tile_dim1 (tile_dim2=0)
    g1[5] = K;                                         // dim0_stride[31:0]
    g1[6] = 0u;                                        // stride hi, dim1_stride lo
    g1[7] = 0u;
    asm volatile("tensor_load_to_lds %0, %1" :: "s"(g0), "s"(g1) : "memory");
}

// ---------------------------------------------------------------------------
// fp32 -> bf16 convert
// ---------------------------------------------------------------------------
__global__ __launch_bounds__(256) void conv_bf16(const float* __restrict__ src,
                                                 unsigned short* __restrict__ dst,
                                                 unsigned int n) {
    unsigned int i = blockIdx.x * 256u + threadIdx.x;
    if (i < n) dst[i] = f2bf(src[i]);
}

// ---------------------------------------------------------------------------
// WMMA GEMM with TDM staging: C[M,N] = A[M,K] * W[N,K]^T.
// Block: 256 threads = 8 waves; block tile 128(M) x 64(N); K-step 64.
// Per K-step every wave issues two tensor_load_to_lds (its 16-row A strip and
// an 8-row slice of the shared 64-row B tile) into the double-buffered LDS
// tile, waits its own TENSORcnt, barriers, then runs 8 WMMAs off ds_load_b128
// fragment fetches. Padded 144B LDS row pitch avoids bank conflicts.
// Fragment layout follows ISA 16-bit A-matrix 16x32:
//   lanes 0-15 : M=lane,    K = 0..7  and 16..23
//   lanes 16-31: M=lane-16, K = 8..15 and 24..31
// ---------------------------------------------------------------------------
#define ROWB 144                           // padded LDS row pitch (bytes)
#define ATILE_B (128 * ROWB)               // A region bytes per buffer
#define BUF_B ((128 + 64) * ROWB)          // one buffer: A(128 rows)+B(64 rows)

template <bool OUT_F32>
__global__ __launch_bounds__(256) void gemm_wmma(const unsigned short* __restrict__ A,
                                                 const unsigned short* __restrict__ W,
                                                 void* __restrict__ Cout,
                                                 int M, int N, int K) {
    __shared__ __align__(16) unsigned char lds[2][BUF_B];   // ~54 KB of 320 KB

    const int lane = threadIdx.x & 31;
    const int wave = threadIdx.x >> 5;
    const int half = lane >> 4;
    const int r    = lane & 15;

    const int nTilesN = N >> 6;
    const int bm = blockIdx.x / nTilesN;
    const int bn = blockIdx.x % nTilesN;
    const int row0 = bm * 128;
    const int col0 = bn * 64;

    const unsigned short* Astrip = A + (size_t)(row0 + wave * 16) * K;
    const unsigned short* Bstrip = W + (size_t)(col0 + wave * 8) * K;
    const unsigned ldsA = (unsigned)(size_t)&lds[0][0] + (unsigned)(wave * 16 * ROWB);
    const unsigned ldsB = (unsigned)(size_t)&lds[0][0] + ATILE_B + (unsigned)(wave * 8 * ROWB);

    v8f acc[4] = {};

    // prologue: stage K-tile 0 into buffer 0
    tdm_load_2d_bf16(ldsA, Astrip, (unsigned)K, 16u, 64u);
    tdm_load_2d_bf16(ldsB, Bstrip, (unsigned)K, 8u, 64u);

    const int nkt = K >> 6;
    for (int kt = 0; kt < nkt; ++kt) {
        const unsigned bufOff = (unsigned)((kt & 1) * BUF_B);
        if (kt + 1 < nkt) {
            const unsigned nOff = (unsigned)(((kt + 1) & 1) * BUF_B);
            const int k1 = (kt + 1) << 6;
            tdm_load_2d_bf16(ldsA + nOff, Astrip + k1, (unsigned)K, 16u, 64u);
            tdm_load_2d_bf16(ldsB + nOff, Bstrip + k1, (unsigned)K, 8u, 64u);
            __builtin_amdgcn_s_wait_tensorcnt((short)2);   // current pair done
        } else {
            __builtin_amdgcn_s_wait_tensorcnt((short)0);
        }
        __syncthreads();                                   // tile published

        const unsigned char* buf  = &lds[0][0] + bufOff;
        const unsigned char* aRow = buf + (wave * 16 + r) * ROWB + half * 16;
        const unsigned char* bRow = buf + ATILE_B + r * ROWB + half * 16;
#pragma unroll
        for (int kk = 0; kk < 2; ++kk) {                   // two 32-K substeps
            Frag a;
            a.u4[0] = *(const uint4*)(aRow + kk * 64);
            a.u4[1] = *(const uint4*)(aRow + kk * 64 + 32);
#pragma unroll
            for (int j = 0; j < 4; ++j) {
                const unsigned char* bp = bRow + j * 16 * ROWB + kk * 64;
                Frag b;
                b.u4[0] = *(const uint4*)(bp);
                b.u4[1] = *(const uint4*)(bp + 32);
                acc[j] = __builtin_amdgcn_wmma_f32_16x16x32_bf16(
                    false, a.v, false, b.v, (short)0, acc[j], false, false);
            }
        }
        __syncthreads();            // reads done before next TDM overwrites
    }

    // C layout: VGPR i of lane -> (row = i + 8*half, col = r) within the tile.
#pragma unroll
    for (int j = 0; j < 4; ++j) {
#pragma unroll
        for (int i = 0; i < 8; ++i) {
            const size_t rr = (size_t)(row0 + wave * 16 + i + 8 * half);
            const size_t cc = (size_t)(col0 + j * 16 + r);
            if (OUT_F32)
                ((float*)Cout)[rr * N + cc] = acc[j][i];
            else
                ((unsigned short*)Cout)[rr * N + cc] = f2bf(acc[j][i]);
        }
    }
}

// ---------------------------------------------------------------------------
// In-place RoPE on a [B,S,nHeads,D] bf16 tensor. One thread per (b,s,h,d<64)
// pair -> race-free and deterministic (buffer is rewritten by GEMM each call).
// ---------------------------------------------------------------------------
__global__ __launch_bounds__(256) void rope_inplace(unsigned short* __restrict__ t,
                                                    int nHeads, unsigned int total) {
    unsigned int idx = blockIdx.x * 256u + threadIdx.x;
    if (idx >= total) return;
    int d = idx & 63;
    unsigned int rest = idx >> 6;
    int h = rest % nHeads; rest /= nHeads;
    int s = rest % S_;
    int b = rest / S_;
    // inv_freq = 10000^(-2d/128) = exp(-d * ln(1e4)/64)
    float ang = (float)s * __expf((float)d * -0.14391156f);
    float sn, cs;
    __sincosf(ang, &sn, &cs);
    size_t base = ((((size_t)b * S_ + s) * nHeads + h) * D_) + d;
    float t1 = bf2f(t[base]);
    float t2 = bf2f(t[base + 64]);
    t[base]      = f2bf(t1 * cs - t2 * sn);
    t[base + 64] = f2bf(t2 * cs + t1 * sn);
}

// ---------------------------------------------------------------------------
// V transpose: [B,S,KV,D] -> [B,KV,D,S] so the P*V B-operand is K-contiguous.
// ---------------------------------------------------------------------------
__global__ __launch_bounds__(256) void transpose_v(const unsigned short* __restrict__ v,
                                                   unsigned short* __restrict__ vt,
                                                   unsigned int total) {
    unsigned int idx = blockIdx.x * 256u + threadIdx.x;
    if (idx >= total) return;
    int d = idx & (D_ - 1);
    unsigned int rest = idx >> 7;
    int kv = rest & (KV_ - 1); rest >>= 3;
    int s = rest & (S_ - 1);
    int b = rest >> 10;
    vt[((((size_t)b * KV_ + kv) * D_) + d) * S_ + s] = v[idx];
}

// ---------------------------------------------------------------------------
// Flash attention (causal, GQA). One wave per (b, head, 16-query tile).
// Scores: 16x32 = Q(16x128) * K^T via 8 wmma; online softmax with shfl_xor
// reductions over 16-lane halves; P goes C-layout -> A-layout through LDS
// (s_wait_dscnt 0 to order the in-wave transpose); O += P(16x32)*V(32x128)
// via 8 wmma with per-row alpha rescale.
// ---------------------------------------------------------------------------
__global__ __launch_bounds__(256) void flash_attn(const unsigned short* __restrict__ q,
                                                  const unsigned short* __restrict__ k,
                                                  const unsigned short* __restrict__ vt,
                                                  unsigned short* __restrict__ attn) {
    __shared__ unsigned short pLds[8][16 * 32];   // per-wave P transpose scratch

    const int lane = threadIdx.x & 31;
    const int wave = threadIdx.x >> 5;
    const int half = lane >> 4;
    const int r    = lane & 15;

    const int task = blockIdx.x * 8 + wave;       // B*H*(S/16) = 8192 tasks
    const int qt = task & 63;
    int rest = task >> 6;
    const int h  = rest & 31;
    const int b  = rest >> 5;
    const int kv = h >> 2;                        // G = 4

    const float scale = 0.08838834764831845f;     // 1/sqrt(128)
    const float NEGB  = -1.0e30f;

    v8f acc[8] = {};
    float m[8], l[8];
#pragma unroll
    for (int i = 0; i < 8; ++i) { m[i] = NEGB; l[i] = 0.0f; }

    const unsigned short* qbase =
        q + ((((size_t)b * S_ + qt * 16 + r) * H_ + h) * D_) + half * 8;
    const unsigned short* vtbase =
        vt + (((size_t)b * KV_ + kv) * D_) * S_ + half * 8;

    const int nk = (qt + 1) * 16;                 // keys needed (causal)
    for (int kt0 = 0; kt0 < nk; kt0 += 32) {
        // ---- scores: two 16x16 C tiles over this 32-key block ----
        v8f sc[2] = {};
#pragma unroll
        for (int kk = 0; kk < D_; kk += 32) {
            Frag aq;
            aq.u4[0] = *(const uint4*)(qbase + kk);
            aq.u4[1] = *(const uint4*)(qbase + kk + 16);
#pragma unroll
            for (int sub = 0; sub < 2; ++sub) {
                const int krow = kt0 + sub * 16 + r;
                const unsigned short* kp =
                    k + ((((size_t)b * S_ + krow) * KV_ + kv) * D_) + half * 8 + kk;
                Frag bk;
                bk.u4[0] = *(const uint4*)(kp);
                bk.u4[1] = *(const uint4*)(kp + 16);
                sc[sub] = __builtin_amdgcn_wmma_f32_16x16x32_bf16(
                    false, aq.v, false, bk.v, (short)0, sc[sub], false, false);
            }
        }
        __builtin_prefetch(k + ((((size_t)b * S_ + kt0 + 32 + r) * KV_ + kv) * D_), 0, 1);

        // ---- mask + online softmax (row i+8*half owned by this half) ----
        float alphaA[8];
#pragma unroll
        for (int i = 0; i < 8; ++i) {
            const int qrow = qt * 16 + i + 8 * half;
            float s0 = sc[0][i] * scale;
            float s1 = sc[1][i] * scale;
            if (kt0 + r      > qrow) s0 = NEGB;
            if (kt0 + 16 + r > qrow) s1 = NEGB;
            float mx = fmaxf(s0, s1);
            mx = fmaxf(mx, __shfl_xor(mx, 1));
            mx = fmaxf(mx, __shfl_xor(mx, 2));
            mx = fmaxf(mx, __shfl_xor(mx, 4));
            mx = fmaxf(mx, __shfl_xor(mx, 8));
            const float mnew = fmaxf(m[i], mx);
            const float p0 = __expf(s0 - mnew);
            const float p1 = __expf(s1 - mnew);
            float rs = p0 + p1;
            rs += __shfl_xor(rs, 1);
            rs += __shfl_xor(rs, 2);
            rs += __shfl_xor(rs, 4);
            rs += __shfl_xor(rs, 8);
            const float alpha = __expf(m[i] - mnew);
            l[i] = l[i] * alpha + rs;
            m[i] = mnew;
            alphaA[i] = alpha;
            pLds[wave][(i + 8 * half) * 32 + r]      = f2bf(p0);
            pLds[wave][(i + 8 * half) * 32 + 16 + r] = f2bf(p1);
        }
        asm volatile("s_wait_dscnt 0" ::: "memory");   // in-wave LDS transpose order

        // ---- P in A-layout from LDS, then O += P * V ----
        Frag pa;
        const unsigned short* pl = &pLds[wave][r * 32 + half * 8];
        pa.u4[0] = *(const uint4*)(pl);
        pa.u4[1] = *(const uint4*)(pl + 16);
#pragma unroll
        for (int j = 0; j < 8; ++j) {
#pragma unroll
            for (int i = 0; i < 8; ++i) acc[j][i] *= alphaA[i];
            const unsigned short* vp = vtbase + (size_t)(j * 16 + r) * S_ + kt0;
            Frag bv;
            bv.u4[0] = *(const uint4*)(vp);
            bv.u4[1] = *(const uint4*)(vp + 16);
            acc[j] = __builtin_amdgcn_wmma_f32_16x16x32_bf16(
                false, pa.v, false, bv.v, (short)0, acc[j], false, false);
        }
        asm volatile("s_wait_dscnt 0" ::: "memory");   // reads done before next tile's stores
    }

    // ---- normalize + store [B,S,H,D] bf16 ----
#pragma unroll
    for (int j = 0; j < 8; ++j) {
#pragma unroll
        for (int i = 0; i < 8; ++i) {
            const int srow = qt * 16 + i + 8 * half;
            const float o = acc[j][i] / l[i];
            attn[((((size_t)b * S_ + srow) * H_ + h) * D_) + j * 16 + r] = f2bf(o);
        }
    }
}

// ---------------------------------------------------------------------------
// Host-side orchestration (all on `stream`, graph-capture safe).
// ---------------------------------------------------------------------------
extern "C" void kernel_launch(void* const* d_in, const int* in_sizes, int n_in,
                              void* d_out, int out_size, void* d_ws, size_t ws_size,
                              hipStream_t stream) {
    (void)in_sizes; (void)n_in; (void)out_size; (void)ws_size;
    const float* x  = (const float*)d_in[0];
    const float* wq = (const float*)d_in[1];
    const float* wk = (const float*)d_in[2];
    const float* wv = (const float*)d_in[3];
    const float* wo = (const float*)d_in[4];
    float* out = (float*)d_out;

    const size_t nXE  = (size_t)M_ * E_;          // 16,777,216
    const size_t nKV  = (size_t)M_ * KV_ * D_;    //  4,194,304

    unsigned short* ws = (unsigned short*)d_ws;
    size_t off = 0;
    auto take = [&](size_t n) { unsigned short* p = ws + off; off += n; return p; };
    unsigned short* xb   = take(nXE);
    unsigned short* wqb  = take(nXE);
    unsigned short* wkb  = take(nKV);   // wk is [1024,4096] = 4,194,304 elems
    unsigned short* wvb  = take(nKV);
    unsigned short* wob  = take(nXE);
    unsigned short* qb   = take(nXE);   // [B,S,H,D], RoPE applied in place
    unsigned short* kb   = take(nKV);   // [B,S,KV,D], RoPE in place
    unsigned short* vb   = take(nKV);   // [B,S,KV,D]
    unsigned short* vtb  = take(nKV);   // [B,KV,D,S]
    unsigned short* attn = take(nXE);   // [B,S,H,D]

    auto cvt = [&](const float* s, unsigned short* d, size_t n) {
        conv_bf16<<<(unsigned)((n + 255) / 256), 256, 0, stream>>>(s, d, (unsigned)n);
    };
    cvt(x,  xb,  nXE);
    cvt(wq, wqb, nXE);
    cvt(wk, wkb, nKV);
    cvt(wv, wvb, nKV);
    cvt(wo, wob, nXE);

    // Projections: grid = (M/128) * (N/64) blocks of 256 threads.
    gemm_wmma<false><<<(M_ / 128) * (E_ / 64), 256, 0, stream>>>(xb, wqb, qb, M_, E_, E_);
    gemm_wmma<false><<<(M_ / 128) * ((KV_ * D_) / 64), 256, 0, stream>>>(xb, wkb, kb, M_, KV_ * D_, E_);
    gemm_wmma<false><<<(M_ / 128) * ((KV_ * D_) / 64), 256, 0, stream>>>(xb, wvb, vb, M_, KV_ * D_, E_);

    // RoPE q and k (in place; one thread per rotation pair).
    {
        unsigned int tq = (unsigned)(B_ * S_ * H_ * 64);
        rope_inplace<<<tq / 256, 256, 0, stream>>>(qb, H_, tq);
        unsigned int tk = (unsigned)(B_ * S_ * KV_ * 64);
        rope_inplace<<<tk / 256, 256, 0, stream>>>(kb, KV_, tk);
    }

    // V transpose for K-contiguous B operand in P*V.
    transpose_v<<<(unsigned)(nKV / 256), 256, 0, stream>>>(vb, vtb, (unsigned)nKV);

    // Flash attention: B*H*(S/16) = 8192 wave-tasks, 8 waves per block.
    flash_attn<<<(B_ * H_ * (S_ / 16)) / 8, 256, 0, stream>>>(qb, kb, vtb, attn);

    // Output projection straight to fp32 d_out.
    gemm_wmma<true><<<(M_ / 128) * (E_ / 64), 256, 0, stream>>>(attn, wob, out, M_, E_, E_);
}
